// SA_82832739270939
// MI455X (gfx1250) — matrix operand order repaired and checked by reading, not compiled, not run on hardware
//
#include <hip/hip_runtime.h>
#include <hip/hip_bf16.h>

// ---------------- Problem constants ----------------
#define BB   16
#define CC   256
#define NH   4
#define DH   64      // CC / NH
#define GG   8
#define CPG  32      // CC / GG
#define NN   1024    // H*W = 32*32
#define EPSV 1e-5f

typedef __attribute__((ext_vector_type(16))) _Float16 v16h;
typedef __attribute__((ext_vector_type(8)))  float    v8f;
typedef __attribute__((ext_vector_type(4)))  float    f4;

typedef __attribute__((ext_vector_type(4))) unsigned u32x4;
typedef __attribute__((ext_vector_type(4))) int      i32x4;
typedef __attribute__((ext_vector_type(8))) int      i32x8;

#if defined(__has_builtin)
#  if __has_builtin(__builtin_amdgcn_tensor_load_to_lds) && \
      __has_builtin(__builtin_amdgcn_s_wait_tensorcnt)
#    define USE_TDM 1
#  endif
#endif

// 16-byte vector of 8 halves for staging copies
struct __align__(16) h8 { _Float16 v[8]; };

// ---------------- WMMA fragment helpers ----------------
// A-matrix (16x32 f16), lane l: row m = l&15; half ho = l>>4.
// halves 0..7  = K (k0 + ho*8 .. +7)      -> 16B contiguous
// halves 8..15 = K (k0 + 16 + ho*8 .. +7) -> 16B contiguous
__device__ __forceinline__ v16h frag_a(const _Float16* base, int ld, int m0, int k0) {
    int lane = threadIdx.x & 31;
    int m = lane & 15, ho = lane >> 4;
    const _Float16* p = base + (size_t)(m0 + m) * ld + k0 + ho * 8;
    union { struct { f4 lo, hi; } f; v16h h; } u;
    u.f.lo = *(const f4*)(p);
    u.f.hi = *(const f4*)(p + 16);
    return u.h;
}

// B-matrix (32x16 f16) stored in LDS TRANSPOSED as [n][k] with leading dim ld.
// lane l: col n = l&15; half ho = l>>4; halves 0..7: K = k0 + ho*8 + e, halves 8..15: K = k0 + 16 + ho*8 + (e-8)
__device__ __forceinline__ v16h frag_b(const _Float16* base, int ld, int n0, int k0) {
    int lane = threadIdx.x & 31;
    int n = lane & 15, ho = lane >> 4;
    const _Float16* p = base + (size_t)(n0 + n) * ld + k0 + ho * 8;
    union { struct { f4 lo, hi; } f; v16h h; } u;
    u.f.lo = *(const f4*)(p);
    u.f.hi = *(const f4*)(p + 16);
    return u.h;
}

__device__ __forceinline__ v8f wmma_f16(v16h a, v16h b, v8f c) {
    return __builtin_amdgcn_wmma_f32_16x16x32_f16(false, a, false, b, (short)0, c,
                                                  false, false);
}

#ifdef USE_TDM
// Issue a TDM 2D tile load: global (row-major, rowStrideElems f16/row) -> LDS.
// tile = tdim0 (contiguous) x tdim1 rows.  Wave-level op; EXEC ignored.
__device__ __forceinline__ void tdm_load_tile_f16(void* ldsDst, const _Float16* gsrc,
                                                  unsigned tdim0, unsigned tdim1,
                                                  unsigned rowStrideElems) {
    unsigned lds_off = (unsigned)(size_t)ldsDst;  // LDS offset = addr[31:0]
    unsigned long long ga = (unsigned long long)(size_t)gsrc;
    u32x4 g0;
    g0.x = 1u;                                           // count=1 (valid user D#)
    g0.y = lds_off;                                      // lds_addr
    g0.z = (unsigned)ga;                                 // global_addr[31:0]
    g0.w = (unsigned)((ga >> 32) & 0x1FFFFFFu) | (2u << 30);  // addr[56:32] | type=2
    i32x8 g1;
    g1[0] = 0x00010000;                // data_size=1 (2 bytes)
    g1[1] = (int)(tdim0 << 16);        // tensor_dim0[15:0] at bits 63:48
    g1[2] = (int)((tdim0 >> 16) | (tdim1 << 16));        // dim0 hi | dim1 lo
    g1[3] = (int)((tdim1 >> 16) | (tdim0 << 16));        // dim1 hi | tile_dim0
    g1[4] = (int)tdim1;                                  // tile_dim1 (tile_dim2=0)
    g1[5] = (int)rowStrideElems;                         // tensor_dim0_stride lo32
    g1[6] = 0;
    g1[7] = 0;
    i32x4 z4 = (i32x4)0;
#if __clang_major__ >= 23
    i32x8 z8 = (i32x8)0;
    __builtin_amdgcn_tensor_load_to_lds(g0, g1, z4, z4, z8, 0);
#else
    __builtin_amdgcn_tensor_load_to_lds(g0, g1, z4, z4, 0);
#endif
}
#endif

// ---------------- Kernel 0: weight f32 -> f16 ----------------
__global__ void cvt_kernel(const float* __restrict__ src, _Float16* __restrict__ dst,
                           int n) {
    int i = blockIdx.x * 256 + threadIdx.x;
    if (i < n) dst[i] = (_Float16)src[i];
}

// ---------------- Kernel 1: GroupNorm -> f16 ----------------
__global__ void gn_kernel(const float* __restrict__ x,
                          const float* __restrict__ w,
                          const float* __restrict__ bias,
                          _Float16* __restrict__ xn) {
    int b = blockIdx.x / GG;
    int g = blockIdx.x % GG;
    const float* xp = x + (size_t)b * CC * NN + (size_t)g * CPG * NN;
    _Float16*    op = xn + (size_t)b * CC * NN + (size_t)g * CPG * NN;
    const int NE = CPG * NN;  // 32768

    __shared__ float r0[256], r1[256];
    int tid = threadIdx.x;
    float s = 0.f, sq = 0.f;
    for (int i = tid; i < NE; i += 256) {
        float v = xp[i];
        s += v; sq += v * v;
    }
    r0[tid] = s; r1[tid] = sq;
    __syncthreads();
    for (int off = 128; off > 0; off >>= 1) {
        if (tid < off) { r0[tid] += r0[tid + off]; r1[tid] += r1[tid + off]; }
        __syncthreads();
    }
    float mean = r0[0] * (1.0f / NE);
    float var  = r1[0] * (1.0f / NE) - mean * mean;
    float inv  = rsqrtf(var + EPSV);
    for (int i = tid; i < NE; i += 256) {
        int cl = i >> 10;  // / NN
        float v = (xp[i] - mean) * inv;
        op[i] = (_Float16)(v * w[g * CPG + cl] + bias[g * CPG + cl]);
    }
}

// ---------------- Kernel 2/4: WMMA GEMM ----------------
// out[b, m, n] = sum_k Wh[m,k] * X[b,k,n] + bias[m]   (K = 256, Nn = 1024)
// outH != null : store f16 (QKV).  else: outF = resid + result (proj + residual).
__global__ void gemm_kernel(const _Float16* __restrict__ Wh,
                            const float* __restrict__ bias,
                            const _Float16* __restrict__ X,
                            _Float16* __restrict__ outH,
                            const float* __restrict__ resid,
                            float* __restrict__ outF,
                            int M) {
    const int K = CC, Nn = NN;
    int b     = blockIdx.z;
    int mBase = blockIdx.y * 128;
    int nBase = blockIdx.x * 64;
    int tid   = threadIdx.x;
    int wave  = tid >> 5;
    int mt    = wave & 3;   // 4 M-subtiles of 32
    int nt    = wave >> 2;  // 2 N-subtiles of 32

    __shared__ __align__(16) _Float16 As[128 * 32];  // [m][k]
    __shared__ __align__(16) _Float16 Bs[64 * 32];   // [n][k] (transposed)

    const _Float16* Xb = X + (size_t)b * K * Nn;
    v8f acc[2][2] = {};

    for (int k0 = 0; k0 < K; k0 += 32) {
        // ---- stage A (128 m x 32 k tile of f16 weights) ----
#ifdef USE_TDM
        if (wave == 0)
            tdm_load_tile_f16(As, Wh + (size_t)mBase * K + k0, 32u, 128u, (unsigned)K);
#else
        {   // vectorized fallback: 1 x b128 per thread
            int r  = tid >> 2;             // 0..63
            int c8 = (tid & 3) * 8;        // 0,8,16,24
            *(h8*)&As[r * 32 + c8] =
                *(const h8*)&Wh[(size_t)(mBase + r) * K + k0 + c8];
        }
#endif
        // ---- stage B (32 k x 64 n) transposed into [n][k] ----
        {   // one b128 global load per thread, 8 x b16 LDS scatter
            int kk = tid >> 3;             // 0..31
            int n8 = (tid & 7) * 8;        // 0..56
            h8 d = *(const h8*)&Xb[(size_t)(k0 + kk) * Nn + nBase + n8];
            #pragma unroll
            for (int e = 0; e < 8; ++e) Bs[(n8 + e) * 32 + kk] = d.v[e];
        }
        if (k0 + 32 < K)  // pull next weight tile toward L2 (global_prefetch_b8)
            __builtin_prefetch(&Wh[(size_t)(mBase + (tid >> 1)) * K + k0 + 32], 0, 3);
#ifdef USE_TDM
        if (wave == 0) __builtin_amdgcn_s_wait_tensorcnt((short)0);
#endif
        __syncthreads();
        #pragma unroll
        for (int im = 0; im < 2; ++im) {
            v16h a = frag_a(As, 32, mt * 32 + im * 16, 0);
            #pragma unroll
            for (int in = 0; in < 2; ++in) {
                v16h bb = frag_b(Bs, 32, nt * 32 + in * 16, 0);
                acc[im][in] = wmma_f16(a, bb, acc[im][in]);
            }
        }
        __syncthreads();
    }

    // Epilogue.  C/D layout: row = r + (lane>>4)*8, col = lane&15
    int lane = tid & 31, ho = lane >> 4, col = lane & 15;
    #pragma unroll
    for (int im = 0; im < 2; ++im)
        #pragma unroll
        for (int in = 0; in < 2; ++in) {
            int nG = nBase + nt * 32 + in * 16 + col;
            #pragma unroll
            for (int r = 0; r < 8; ++r) {
                int mG = mBase + mt * 32 + im * 16 + ho * 8 + r;
                float vres = acc[im][in][r] + bias[mG];
                size_t o = (size_t)b * M * Nn + (size_t)mG * Nn + nG;
                if (outH) outH[o] = (_Float16)vres;
                else      outF[o] = resid[o] + vres;
            }
        }
}

// ---------------- Kernel 3: fused attention ----------------
// One block per (b, h, 32-row i-tile).  Scores live entirely in LDS (f16).
#define ATT_QT_BYTES   (32 * 64 * 2)        // 4 KB   q^T [i][c]
#define ATT_S_BYTES    (32 * 1024 * 2)      // 64 KB  S   [i][j] f16
#define ATT_STG_BYTES  (64 * 64 * 2)        // 8 KB   staging tile
#define ATT_SMEM       (ATT_QT_BYTES + ATT_S_BYTES + ATT_STG_BYTES)

__global__ void attn_kernel(const _Float16* __restrict__ qkv,
                            _Float16* __restrict__ ovec) {
    extern __shared__ char smem[];
    _Float16* qT  = (_Float16*)smem;                                // [32][64]
    _Float16* S   = (_Float16*)(smem + ATT_QT_BYTES);               // [32][1024]
    _Float16* stg = (_Float16*)(smem + ATT_QT_BYTES + ATT_S_BYTES); // [64][64]

    int iBase = blockIdx.x * 32;
    int h = blockIdx.y, b = blockIdx.z;
    int tid = threadIdx.x;
    int wave = tid >> 5;
    int mt = wave & 1;        // 2 i-subtiles of 16
    int nt = wave >> 1;       // 4 col-subtiles of 16
    const float sc = 0.125f;  // dh^-0.5

    const size_t base = (size_t)b * 3 * CC * NN;
    const _Float16* q = qkv + base + (size_t)(h * DH) * NN;
    const _Float16* k = qkv + base + (size_t)(CC + h * DH) * NN;
    const _Float16* v = qkv + base + (size_t)(2 * CC + h * DH) * NN;

    // Load q^T tile [i][c]: b128 global reads, b16 LDS scatter
    {
        int cc  = tid >> 2;            // 0..63
        int ii8 = (tid & 3) * 8;       // 0,8,16,24
        h8 d = *(const h8*)&q[(size_t)cc * NN + iBase + ii8];
        #pragma unroll
        for (int e = 0; e < 8; ++e) qT[(ii8 + e) * 64 + cc] = d.v[e];
    }

    // ---- Pass 1: S = (q^T k) * sc ----
    for (int jb = 0; jb < 16; ++jb) {
        __syncthreads();
        // stage k^T [j][c]: 2 x (b128 read + 8 b16 scatter) per thread
        #pragma unroll
        for (int i = 0; i < 2; ++i) {
            int lin = tid + i * 256;       // 0..511
            int cc  = lin >> 3;            // 0..63
            int jj8 = (lin & 7) * 8;       // 0..56
            h8 d = *(const h8*)&k[(size_t)cc * NN + jb * 64 + jj8];
            #pragma unroll
            for (int e = 0; e < 8; ++e) stg[(jj8 + e) * 64 + cc] = d.v[e];
        }
        __syncthreads();
        v8f acc = {};
        #pragma unroll
        for (int kk = 0; kk < 64; kk += 32) {
            v16h a  = frag_a(qT, 64, mt * 16, kk);
            v16h bb = frag_b(stg, 64, nt * 16, kk);
            acc = wmma_f16(a, bb, acc);
        }
        int lane = tid & 31, ho = lane >> 4, col = lane & 15;
        #pragma unroll
        for (int r = 0; r < 8; ++r) {
            int row = mt * 16 + ho * 8 + r;
            S[row * 1024 + jb * 64 + nt * 16 + col] = (_Float16)(acc[r] * sc);
        }
    }
    __syncthreads();

    // ---- Pass 2: in-place softmax, 8 lanes per row ----
    {
        int row = tid >> 3, ls = tid & 7;
        float m = -1e30f;
        for (int c0 = ls; c0 < 1024; c0 += 8) m = fmaxf(m, (float)S[row * 1024 + c0]);
        #pragma unroll
        for (int off = 4; off; off >>= 1) m = fmaxf(m, __shfl_xor(m, off, 8));
        float sum = 0.f;
        for (int c0 = ls; c0 < 1024; c0 += 8) sum += __expf((float)S[row * 1024 + c0] - m);
        #pragma unroll
        for (int off = 4; off; off >>= 1) sum += __shfl_xor(sum, off, 8);
        float inv = 1.0f / sum;
        for (int c0 = ls; c0 < 1024; c0 += 8)
            S[row * 1024 + c0] = (_Float16)(__expf((float)S[row * 1024 + c0] - m) * inv);
    }
    __syncthreads();

    // ---- Pass 3: O^T[i][c] = A[i][j] * V^T[j][c] ----
    v8f acc = {};
    for (int jb = 0; jb < 16; ++jb) {
        __syncthreads();
        // stage v [c][j] — already the [n][k] layout; b128 on both sides
        #pragma unroll
        for (int i = 0; i < 2; ++i) {
            int lin = tid + i * 256;
            int cc  = lin >> 3;
            int jj8 = (lin & 7) * 8;
            *(h8*)&stg[cc * 64 + jj8] =
                *(const h8*)&v[(size_t)cc * NN + jb * 64 + jj8];
        }
        __syncthreads();
        #pragma unroll
        for (int kk = 0; kk < 64; kk += 32) {
            v16h a  = frag_a(S, 1024, mt * 16, jb * 64 + kk);
            v16h bb = frag_b(stg, 64, nt * 16, kk);
            acc = wmma_f16(a, bb, acc);
        }
    }
    // store transposed back to [C][N] layout for the projection GEMM
    int lane = tid & 31, ho = lane >> 4, col = lane & 15;
    #pragma unroll
    for (int r = 0; r < 8; ++r) {
        int iL = mt * 16 + ho * 8 + r;
        int cL = nt * 16 + col;
        ovec[(size_t)b * CC * NN + (size_t)(h * DH + cL) * NN + iBase + iL] =
            (_Float16)acc[r];
    }
}

// ---------------- Host launcher ----------------
extern "C" void kernel_launch(void* const* d_in, const int* in_sizes, int n_in,
                              void* d_out, int out_size, void* d_ws, size_t ws_size,
                              hipStream_t stream) {
    (void)in_sizes; (void)n_in; (void)out_size; (void)ws_size;
    const float* x    = (const float*)d_in[0];
    const float* nw   = (const float*)d_in[1];
    const float* nb   = (const float*)d_in[2];
    const float* qkvw = (const float*)d_in[3];
    const float* qkvb = (const float*)d_in[4];
    const float* pw   = (const float*)d_in[5];
    const float* pb   = (const float*)d_in[6];
    float* out = (float*)d_out;

    char* ws = (char*)d_ws;
    size_t off = 0;
    _Float16* xn    = (_Float16*)(ws + off); off += (size_t)BB * CC * NN * 2;      //  8 MB
    _Float16* qkvh  = (_Float16*)(ws + off); off += (size_t)BB * 3 * CC * NN * 2;  // 24 MB
    _Float16* oh    = (_Float16*)(ws + off); off += (size_t)BB * CC * NN * 2;      //  8 MB
    _Float16* qkvwh = (_Float16*)(ws + off); off += (size_t)3 * CC * CC * 2;       // 384 KB
    _Float16* pwh   = (_Float16*)(ws + off); off += (size_t)CC * CC * 2;           // 128 KB

    // 0) one-time weight conversion f32 -> f16
    cvt_kernel<<<dim3((3 * CC * CC + 255) / 256), 256, 0, stream>>>(qkvw, qkvwh,
                                                                    3 * CC * CC);
    cvt_kernel<<<dim3((CC * CC + 255) / 256), 256, 0, stream>>>(pw, pwh, CC * CC);

    // 1) GroupNorm -> f16 activations
    gn_kernel<<<dim3(BB * GG), 256, 0, stream>>>(x, nw, nb, xn);

    // 2) QKV projection: [768x256] x [256x1024] per batch (WMMA + TDM staging)
    gemm_kernel<<<dim3(NN / 64, (3 * CC) / 128, BB), 256, 0, stream>>>(
        qkvwh, qkvb, xn, qkvh, nullptr, nullptr, 3 * CC);

    // 3) Fused attention (scores entirely in LDS)
    hipFuncSetAttribute((const void*)attn_kernel,
                        hipFuncAttributeMaxDynamicSharedMemorySize, ATT_SMEM);
    attn_kernel<<<dim3(NN / 32, NH, BB), 256, ATT_SMEM, stream>>>(qkvh, oh);

    // 4) Output projection + residual (WMMA, f32 out)
    gemm_kernel<<<dim3(NN / 64, CC / 128, BB), 256, 0, stream>>>(
        pwh, pb, oh, nullptr, x, out, CC);
}
